// MiMoMoeAttention_3728031613184
// MI455X (gfx1250) — compile-verified, matrix operand order checked by reading, not run on hardware
//
#include <hip/hip_runtime.h>

// Problem constants (match reference)
#define S_LEN 2048
#define HID   4096
#define NH    32
#define NKV   8
#define HD    128
#define WINSZ 1024

typedef __attribute__((ext_vector_type(16))) __bf16 v16bf;
typedef __attribute__((ext_vector_type(8)))  __bf16 v8bf;
typedef __attribute__((ext_vector_type(8)))  float  v8f;

__device__ __forceinline__ unsigned short f2bf(float f) {
  unsigned u = __builtin_bit_cast(unsigned, f);
  u += 0x7FFFu + ((u >> 16) & 1u);            // round-to-nearest-even
  return (unsigned short)(u >> 16);
}

// Load a 16-element bf16 WMMA fragment for lane (r, hi): two contiguous
// 8-element (16-byte) runs at p and p+16 (ISA 16-bit A/B lane layout).
__device__ __forceinline__ v16bf load_frag(const unsigned short* p) {
  v8bf a = *(const v8bf*)(p);
  v8bf b = *(const v8bf*)(p + 16);
  return __builtin_shufflevector(a, b, 0,1,2,3,4,5,6,7,8,9,10,11,12,13,14,15);
}

__device__ __forceinline__ v8f wmma_bf16(v16bf a, v16bf b, v8f c) {
  return __builtin_amdgcn_wmma_f32_16x16x32_bf16(false, a, false, b, (short)0, c,
                                                 false, false);
}

// ---------------------------------------------------------------- conversions
__global__ void cvt_f32_bf16(const float* __restrict__ x,
                             unsigned short* __restrict__ y, int n) {
  int i = (blockIdx.x * blockDim.x + threadIdx.x) * 4;
  if (i + 3 < n) {
    float4 v = *(const float4*)(x + i);
    ushort4 o = { f2bf(v.x), f2bf(v.y), f2bf(v.z), f2bf(v.w) };
    *(ushort4*)(y + i) = o;
  }
}

// Tiled transpose: W[k][n] fp32 -> Wt[n][k] bf16, 32x32 LDS tiles.
// blockDim = 256 (32x8). Coalesced reads and writes.
__global__ void __launch_bounds__(256)
transpose_bf16(const float* __restrict__ W, unsigned short* __restrict__ Wt,
               int K, int N) {
  __shared__ unsigned short tile[32][33];
  int tx = threadIdx.x & 31, ty = threadIdx.x >> 5;
  int nb = blockIdx.x * 32, kb = blockIdx.y * 32;
#pragma unroll
  for (int j = 0; j < 32; j += 8)
    tile[ty + j][tx] = f2bf(W[(size_t)(kb + ty + j) * N + nb + tx]);
  __syncthreads();
#pragma unroll
  for (int j = 0; j < 32; j += 8)
    Wt[(size_t)(nb + ty + j) * K + kb + tx] = tile[tx][ty + j];
}

// NeoX RoPE on fp32 [S][nh][HD], emit bf16 [S][nh*HD]
__global__ void rope_bf16(const float* __restrict__ x, const int* __restrict__ pos,
                          unsigned short* __restrict__ out, int nh) {
  int i = blockIdx.x * blockDim.x + threadIdx.x;
  int total = S_LEN * nh * 64;
  if (i >= total) return;
  int d = i % 64, t = i / 64, h = t % nh, s = t / nh;
  float inv = expf(-(float)d * (13.815510558f / 64.f));   // THETA^-(d/64), ln(1e6)
  float ang = (float)pos[s] * inv;
  float sn, cs; sincosf(ang, &sn, &cs);
  const float* xr = x + ((size_t)s * nh + h) * HD;
  float x1 = xr[d], x2 = xr[d + 64];
  unsigned short* o = out + ((size_t)s * nh + h) * HD;
  o[d]      = f2bf(x1 * cs - x2 * sn);
  o[d + 64] = f2bf(x2 * cs + x1 * sn);
}

// ------------------------------------------------------------------- WMMA GEMM
// C[M][N] fp32 = A[M][K] bf16 (row-major) @ Bt[N][K] bf16 (pre-transposed) + bias
// 128 threads = 4 waves; each wave computes a 16x64 tile (A frag reused x4).
// Ping-pong double buffering (2x unrolled k-loop): each buffer is refilled in
// place right after its WMMAs consume it, so no register-copy rotation.
// Requires K % 64 == 0 and K >= 128 (true for all calls here: K = 4096).
__global__ void __launch_bounds__(128)
gemm_bf16(const unsigned short* __restrict__ A, const unsigned short* __restrict__ Bt,
          const float* __restrict__ bias, float* __restrict__ C,
          int M, int N, int K) {
  int lane = threadIdx.x & 31, w = threadIdx.x >> 5;
  int r = lane & 15, hi = lane >> 4;
  int m0 = blockIdx.y * 16;
  int n0 = (blockIdx.x * 4 + w) * 64;

  v8f acc[4];
#pragma unroll
  for (int t = 0; t < 4; ++t) acc[t] = (v8f){0.f,0.f,0.f,0.f,0.f,0.f,0.f,0.f};

  const unsigned short* ap = A  + (size_t)(m0 + r) * K + 8 * hi;
  const unsigned short* bp = Bt + (size_t)(n0 + r) * K + 8 * hi;

  // prologue: buffer0 <- step 0, buffer1 <- step 32
  v16bf af0 = load_frag(ap), af1 = load_frag(ap + 32);
  v16bf bf0[4], bf1[4];
#pragma unroll
  for (int t = 0; t < 4; ++t) bf0[t] = load_frag(bp + (size_t)t * 16 * K);
#pragma unroll
  for (int t = 0; t < 4; ++t) bf1[t] = load_frag(bp + 32 + (size_t)t * 16 * K);

  for (int k0 = 0; k0 < K - 96; k0 += 64) {
    // consume buffer0 (step k0), refill with step k0+64
#pragma unroll
    for (int t = 0; t < 4; ++t) acc[t] = wmma_bf16(af0, bf0[t], acc[t]);
    af0 = load_frag(ap + k0 + 64);
#pragma unroll
    for (int t = 0; t < 4; ++t)
      bf0[t] = load_frag(bp + k0 + 64 + (size_t)t * 16 * K);
    // consume buffer1 (step k0+32), refill with step k0+96
#pragma unroll
    for (int t = 0; t < 4; ++t) acc[t] = wmma_bf16(af1, bf1[t], acc[t]);
    af1 = load_frag(ap + k0 + 96);
#pragma unroll
    for (int t = 0; t < 4; ++t)
      bf1[t] = load_frag(bp + k0 + 96 + (size_t)t * 16 * K);
  }
  // tail: buffers hold steps K-64 and K-32
#pragma unroll
  for (int t = 0; t < 4; ++t) acc[t] = wmma_bf16(af0, bf0[t], acc[t]);
#pragma unroll
  for (int t = 0; t < 4; ++t) acc[t] = wmma_bf16(af1, bf1[t], acc[t]);

#pragma unroll
  for (int t = 0; t < 4; ++t) {
    int n = n0 + t * 16 + r;
    float bv = bias ? bias[n] : 0.f;
#pragma unroll
    for (int i = 0; i < 8; ++i)
      C[(size_t)(m0 + i + 8 * hi) * N + n] = acc[t][i] + bv;
  }
}

// ------------------------------------------------------------ flash attention
// One wave per (head h, 16-row query tile). KV tiles of 32 keys.
// Sink: init m = sink[h], l = 1 (the sink's softmax term), acc = 0.
__global__ void __launch_bounds__(256)
attn_fwd(const unsigned short* __restrict__ qb,   // [S][NH*HD] bf16 (post-RoPE)
         const unsigned short* __restrict__ kb,   // [S][NKV*HD] bf16 (post-RoPE)
         const unsigned short* __restrict__ vt,   // [NKV*HD][S] bf16 (transposed)
         const float* __restrict__ sink,
         unsigned short* __restrict__ ctx) {      // [S][NH*HD] bf16
  __shared__ unsigned short plds[8 * 16 * 32];    // per-wave 16x32 P staging
  int lane = threadIdx.x & 31, w = threadIdx.x >> 5;
  int r = lane & 15, hi = lane >> 4;
  int h = blockIdx.x, hk = h >> 2;
  int q0 = (blockIdx.y * 8 + w) * 16;
  const float scale = 0.088388347648318447f;      // 1/sqrt(128)

  // Preload Q fragments for all 4 k-steps (D=128)
  v16bf qf[4];
  const unsigned short* qrow = qb + (size_t)(q0 + r) * (NH * HD) + h * HD + 8 * hi;
#pragma unroll
  for (int ks = 0; ks < 4; ++ks) qf[ks] = load_frag(qrow + ks * 32);

  v8f acc[8];
#pragma unroll
  for (int t = 0; t < 8; ++t) acc[t] = (v8f){0.f,0.f,0.f,0.f,0.f,0.f,0.f,0.f};
  float m_[8], l_[8];
  float snk = sink[h];
#pragma unroll
  for (int i = 0; i < 8; ++i) { m_[i] = snk; l_[i] = 1.f; }

  int jlo = q0 - WINSZ + 1; if (jlo < 0) jlo = 0;
  int kt0 = jlo >> 5, kt1 = (q0 + 15) >> 5;
  unsigned short* pw = plds + w * 512;

  for (int kt = kt0; kt <= kt1; ++kt) {
    int jb = kt * 32;

    // hint-prefetch next KV tile (global_prefetch_b8)
    if (kt < kt1) {
      const unsigned short* nk =
          kb + (size_t)(jb + 32 + r) * (NKV * HD) + hk * HD;
      __builtin_prefetch(nk, 0, 3);
      __builtin_prefetch(nk + (size_t)16 * (NKV * HD), 0, 3);
      __builtin_prefetch(vt + (size_t)(hk * HD + r * 8) * S_LEN + jb + 32, 0, 3);
    }

    // Load all 8 K fragments first (clause together), then run the WMMA chains.
    const unsigned short* kb0 =
        kb + (size_t)(jb + r) * (NKV * HD) + hk * HD + 8 * hi;
    const unsigned short* kb1 = kb0 + (size_t)16 * (NKV * HD);
    v16bf kf0[4], kf1[4];
#pragma unroll
    for (int ks = 0; ks < 4; ++ks) kf0[ks] = load_frag(kb0 + ks * 32);
#pragma unroll
    for (int ks = 0; ks < 4; ++ks) kf1[ks] = load_frag(kb1 + ks * 32);

    v8f s0 = (v8f){0.f,0.f,0.f,0.f,0.f,0.f,0.f,0.f};
    v8f s1 = s0;
#pragma unroll
    for (int ks = 0; ks < 4; ++ks) {        // s0/s1 chains alternate for overlap
      s0 = wmma_bf16(qf[ks], kf0[ks], s0);
      s1 = wmma_bf16(qf[ks], kf1[ks], s1);
    }

    float p0[8], p1[8];
#pragma unroll
    for (int i = 0; i < 8; ++i) {
      int row = q0 + i + 8 * hi;
      int c0 = jb + r, c1 = c0 + 16;
      bool ok0 = (c0 <= row) && (c0 > row - WINSZ);
      bool ok1 = (c1 <= row) && (c1 > row - WINSZ);
      float v0 = ok0 ? s0[i] * scale : -1e30f;
      float v1 = ok1 ? s1[i] * scale : -1e30f;
      float mx = fmaxf(v0, v1);
#pragma unroll
      for (int d = 1; d < 16; d <<= 1) mx = fmaxf(mx, __shfl_xor(mx, d, 32));
      float mn = fmaxf(m_[i], mx);
      float alpha = __expf(m_[i] - mn);
      float e0 = ok0 ? __expf(v0 - mn) : 0.f;
      float e1 = ok1 ? __expf(v1 - mn) : 0.f;
      float sum = e0 + e1;
#pragma unroll
      for (int d = 1; d < 16; d <<= 1) sum += __shfl_xor(sum, d, 32);
      l_[i] = l_[i] * alpha + sum;
      m_[i] = mn;
      p0[i] = e0; p1[i] = e1;
#pragma unroll
      for (int t = 0; t < 8; ++t) acc[t][i] *= alpha;
    }

    // Stage P (16x32 bf16) in this wave's LDS slice; re-read as A-fragment.
#pragma unroll
    for (int i = 0; i < 8; ++i) {
      int row = i + 8 * hi;
      pw[row * 32 + r]      = f2bf(p0[i]);
      pw[row * 32 + r + 16] = f2bf(p1[i]);
    }
    asm volatile("s_wait_dscnt 0" ::: "memory");  // per-wave LDS RAW fence
    v16bf pf = load_frag(pw + r * 32 + 8 * hi);

    // P x V: grouped loads of 4 fragments, then 4 independent WMMA chains.
    const unsigned short* vb0 = vt + (size_t)(hk * HD + r) * S_LEN + jb + 8 * hi;
#pragma unroll
    for (int g = 0; g < 2; ++g) {
      v16bf vfr[4];
#pragma unroll
      for (int u = 0; u < 4; ++u)
        vfr[u] = load_frag(vb0 + (size_t)((g * 4 + u) * 16) * S_LEN);
#pragma unroll
      for (int u = 0; u < 4; ++u)
        acc[g * 4 + u] = wmma_bf16(pf, vfr[u], acc[g * 4 + u]);
    }
  }

#pragma unroll
  for (int t = 0; t < 8; ++t) {
    int d = t * 16 + r;
#pragma unroll
    for (int i = 0; i < 8; ++i) {
      int row = q0 + i + 8 * hi;
      ctx[(size_t)row * (NH * HD) + h * HD + d] = f2bf(acc[t][i] / l_[i]);
    }
  }
}

// ---------------------------------------------------------------------- launch
extern "C" void kernel_launch(void* const* d_in, const int* in_sizes, int n_in,
                              void* d_out, int out_size, void* d_ws, size_t ws_size,
                              hipStream_t stream) {
  const int*   pos  = (const int*)  d_in[0];
  const float* hs   = (const float*)d_in[1];
  const float* Wq   = (const float*)d_in[2];
  const float* bq   = (const float*)d_in[3];
  const float* Wk   = (const float*)d_in[4];
  const float* bk   = (const float*)d_in[5];
  const float* Wv   = (const float*)d_in[6];
  const float* bv   = (const float*)d_in[7];
  const float* Wo   = (const float*)d_in[8];
  const float* sink = (const float*)d_in[9];
  float* out = (float*)d_out;

  // Workspace layout (bytes). Total ~185 MiB.
  char* p = (char*)d_ws;
  unsigned short* hsb = (unsigned short*)(p);                       // 16 MiB
  unsigned short* Wqt = (unsigned short*)(p + 16777216);            // 32 MiB
  unsigned short* Wkt = (unsigned short*)(p + 50331648);            //  8 MiB
  unsigned short* Wvt = (unsigned short*)(p + 58720256);            //  8 MiB
  unsigned short* Wot = (unsigned short*)(p + 67108864);            // 32 MiB
  float*          qf  = (float*)        (p + 100663296);           // 32 MiB
  float*          kf  = (float*)        (p + 134217728);           //  8 MiB
  float*          vf  = (float*)        (p + 142606336);           //  8 MiB
  unsigned short* qb  = (unsigned short*)(p + 150994944);          // 16 MiB
  unsigned short* kb  = (unsigned short*)(p + 167772160);          //  4 MiB
  unsigned short* vtb = (unsigned short*)(p + 171966464);          //  4 MiB
  unsigned short* ctx = (unsigned short*)(p + 176160768);          // 16 MiB

  // 1) hidden states -> bf16 (vectorized x4)
  cvt_f32_bf16<<<(S_LEN * HID) / (256 * 4), 256, 0, stream>>>(hs, hsb, S_LEN * HID);

  // 2) weights -> bf16, transposed to [N][K] (tiled, coalesced)
  transpose_bf16<<<dim3((NH  * HD) / 32, HID / 32), 256, 0, stream>>>(Wq, Wqt, HID, NH  * HD);
  transpose_bf16<<<dim3((NKV * HD) / 32, HID / 32), 256, 0, stream>>>(Wk, Wkt, HID, NKV * HD);
  transpose_bf16<<<dim3((NKV * HD) / 32, HID / 32), 256, 0, stream>>>(Wv, Wvt, HID, NKV * HD);
  transpose_bf16<<<dim3(HID / 32, (NH * HD) / 32), 256, 0, stream>>>(Wo, Wot, NH * HD, HID);

  // 3) QKV projections (bf16 WMMA, fp32 out)
  gemm_bf16<<<dim3((NH  * HD) / 256, S_LEN / 16), 128, 0, stream>>>(
      hsb, Wqt, bq, qf, S_LEN, NH  * HD, HID);
  gemm_bf16<<<dim3((NKV * HD) / 256, S_LEN / 16), 128, 0, stream>>>(
      hsb, Wkt, bk, kf, S_LEN, NKV * HD, HID);
  gemm_bf16<<<dim3((NKV * HD) / 256, S_LEN / 16), 128, 0, stream>>>(
      hsb, Wvt, bv, vf, S_LEN, NKV * HD, HID);

  // 4) RoPE (q, k) -> bf16; V -> transposed bf16 (same tiled transpose)
  rope_bf16<<<(S_LEN * NH  * 64) / 256, 256, 0, stream>>>(qf, pos, qb, NH);
  rope_bf16<<<(S_LEN * NKV * 64) / 256, 256, 0, stream>>>(kf, pos, kb, NKV);
  transpose_bf16<<<dim3((NKV * HD) / 32, S_LEN / 32), 256, 0, stream>>>(
      vf, vtb, S_LEN, NKV * HD);

  // 5) sliding-window GQA flash attention with sink
  attn_fwd<<<dim3(NH, S_LEN / 128), 256, 0, stream>>>(qb, kb, vtb, sink, ctx);

  // 6) output projection
  gemm_bf16<<<dim3(HID / 256, S_LEN / 16), 128, 0, stream>>>(
      ctx, Wot, nullptr, out, S_LEN, HID, NH * HD);
}
